// CAMILSelfAttention_49091476193808
// MI455X (gfx1250) — matrix operand order repaired and checked by reading, not compiled, not run on hardware
//
#include <hip/hip_runtime.h>

// ---------------------------------------------------------------------------
// CAMIL self-attention for MI455X (gfx1250, wave32, WMMA).
//
//   qk = x @ W_qk^T ; q,k = split(qk) ; v = x @ W_v^T
//   w  = inv_scale * rowdot(q, adj @ k) ; w = softmax_N(w) ; out = w * v
//
// adj (268 MB fp32) is the traffic floor (~11.5us @ 23.3 TB/s); it is read
// exactly once and converted to bf16 in registers (0/1 exact). k is kept as
// bf16 hi + residual lo so adj@k_hi + adj@k_lo ~ fp32. The adj GEMM is
// K-split 4 ways (deterministic partials, no atomics). Both GEMMs ping/pong
// two register tile buffers (manual 2x unroll, no rotation copies) and pin
// the schedule (VMEM group -> WMMA group) with sched_group_barrier.
// ---------------------------------------------------------------------------

typedef __attribute__((ext_vector_type(16))) __bf16 v16bf;
typedef __attribute__((ext_vector_type(8)))  __bf16 v8bf;
typedef __attribute__((ext_vector_type(4)))  __bf16 v4bf;
typedef __attribute__((ext_vector_type(8)))  float  v8f;
typedef __attribute__((ext_vector_type(4)))  float  v4f;

#define BATCH  4
#define NPTS   4096
#define DIN    512
#define ADIM   64
#define BN     (BATCH * NPTS)
#define KSPLIT 4

#if defined(__has_builtin)
#if __has_builtin(__builtin_amdgcn_sched_group_barrier)
#define SCHED_GROUP(mask, n, id) __builtin_amdgcn_sched_group_barrier(mask, n, id)
#else
#define SCHED_GROUP(mask, n, id)
#endif
#else
#define SCHED_GROUP(mask, n, id)
#endif
// sched_group_barrier masks: 0x20 = VMEM read, 0x8 = MFMA/WMMA

// B-fragment (32x16 bf16): lane col = lane%16, K = base + (lane/16)*8 + {0..7, 16..23}
__device__ __forceinline__ v16bf ld_bfrag(const __bf16* __restrict__ p) {
  v8bf b0 = *(const v8bf*)p;
  v8bf b1 = *(const v8bf*)(p + 16);
  v16bf r;
#pragma unroll
  for (int i = 0; i < 8; ++i) { r[i] = b0[i]; r[i + 8] = b1[i]; }
  return r;
}

__device__ __forceinline__ v8f wmma_bf16(v16bf a, v16bf b, v8f c) {
  return __builtin_amdgcn_wmma_f32_16x16x32_bf16(false, a, false, b, (short)0,
                                                 c, false, false);
}

// ---------------------------------------------------------------- fp32->bf16
__global__ __launch_bounds__(256)
void cvt_f32_to_bf16_x4(const float* __restrict__ s, __bf16* __restrict__ d, int n4) {
  int i = blockIdx.x * blockDim.x + threadIdx.x;
  if (i >= n4) return;
  v4f f = *(const v4f*)(s + 4 * (size_t)i);
  v4bf b;
#pragma unroll
  for (int j = 0; j < 4; ++j) b[j] = (__bf16)f[j];
  *(v4bf*)(d + 4 * (size_t)i) = b;
}

// ------------------------------------------------- projection GEMM (x @ W^T)
struct PTile { v8bf alo, ahi; v16bf bfr[4]; };

__device__ __forceinline__ void p_load(PTile& t, const __bf16* __restrict__ arow,
                                       const __bf16* __restrict__ wrow, int kt) {
  t.alo = *(const v8bf*)(arow + kt);
  t.ahi = *(const v8bf*)(arow + kt + 16);
#pragma unroll
  for (int c = 0; c < 4; ++c) t.bfr[c] = ld_bfrag(wrow + (size_t)c * 16 * DIN + kt);
}

__device__ __forceinline__ void p_compute(const PTile& t, v8f acc[4]) {
  v16bf a;
#pragma unroll
  for (int i = 0; i < 8; ++i) { a[i] = t.alo[i]; a[i + 8] = t.ahi[i]; }
#pragma unroll
  for (int c = 0; c < 4; ++c) acc[c] = wmma_bf16(a, t.bfr[c], acc[c]);
  SCHED_GROUP(0x0020, 10, 0);   // 10 VMEM reads first ...
  SCHED_GROUP(0x0008, 4, 0);    // ... then 4 WMMAs
}

// mode 0: write fp32 outV[row*E + e]
// mode 1: e<64 -> q fp32 [row][e] ; e>=64 -> kT hi/lo bf16 [b][e-64][n]
__global__ __launch_bounds__(128)
void gemm_proj(const __bf16* __restrict__ X,   // [BN][DIN] bf16
               const __bf16* __restrict__ W,   // [E][DIN]  bf16
               float* __restrict__ outQ,
               __bf16* __restrict__ kThi, __bf16* __restrict__ kTlo,
               float* __restrict__ outV, int E, int mode) {
  const int lane = threadIdx.x & 31;
  const int wv   = threadIdx.x >> 5;
  const int csub = lane & 15;
  const int half = lane >> 4;
  const int m0 = blockIdx.x * 64 + wv * 16;
  const int e0 = blockIdx.y * 64;
  const int mrow  = m0 + csub;
  const int koffA = half * 8;

  const __bf16* arow = X + (size_t)mrow * DIN + koffA;
  const __bf16* wrow = W + (size_t)(e0 + csub) * DIN + half * 8;

  v8f acc[4] = {};
  PTile tA, tB;
  p_load(tA, arow, wrow, 0);
  // DIN/32 = 16 tiles -> 8 ping/pong pairs, no rotation copies
  for (int kt = 0; kt < DIN; kt += 64) {
    p_load(tB, arow, wrow, kt + 32);
    p_compute(tA, acc);
    const int ktn = (kt + 64 < DIN) ? (kt + 64) : 0;
    p_load(tA, arow, wrow, ktn);
    p_compute(tB, acc);
  }

#pragma unroll
  for (int c = 0; c < 4; ++c) {
    const int e = e0 + c * 16 + csub;
#pragma unroll
    for (int r = 0; r < 8; ++r) {
      const int row = m0 + r + half * 8;       // C layout: vgpr r, half -> M
      const float val = acc[c][r];
      if (mode == 0) {
        outV[(size_t)row * E + e] = val;
      } else if (e < ADIM) {
        outQ[(size_t)row * ADIM + e] = val;
      } else {
        const int b = row >> 12, nn = row & (NPTS - 1), ee = e - ADIM;
        const __bf16 hi = (__bf16)val;
        const float rem = val - (float)hi;
        kThi[((size_t)b * ADIM + ee) * NPTS + nn] = hi;
        kTlo[((size_t)b * ADIM + ee) * NPTS + nn] = (__bf16)rem;
      }
    }
  }
}

// --------------------- partial kagg = adj @ k (hi+lo), wpart = q . kagg / 8
struct ATile { v4f fa[4]; v16bf bh[4], bl[4]; };

__device__ __forceinline__ void a_load(ATile& t, const float* __restrict__ arow,
                                       const __bf16* __restrict__ khrow,
                                       const __bf16* __restrict__ klrow, int kt) {
  t.fa[0] = *(const v4f*)(arow + kt);
  t.fa[1] = *(const v4f*)(arow + kt + 4);
  t.fa[2] = *(const v4f*)(arow + kt + 16);
  t.fa[3] = *(const v4f*)(arow + kt + 20);
#pragma unroll
  for (int c = 0; c < 4; ++c) {
    t.bh[c] = ld_bfrag(khrow + (size_t)c * 16 * NPTS + kt);
    t.bl[c] = ld_bfrag(klrow + (size_t)c * 16 * NPTS + kt);
  }
}

__device__ __forceinline__ void a_compute(const ATile& t, v8f acc[4]) {
  v16bf a;                                     // adj 0/1 exact in bf16
#pragma unroll
  for (int i = 0; i < 4; ++i) {
    a[i]      = (__bf16)t.fa[0][i];
    a[i + 4]  = (__bf16)t.fa[1][i];
    a[i + 8]  = (__bf16)t.fa[2][i];
    a[i + 12] = (__bf16)t.fa[3][i];
  }
#pragma unroll
  for (int c = 0; c < 4; ++c) {                // adj fragment reused 8x
    acc[c] = wmma_bf16(a, t.bh[c], acc[c]);
    acc[c] = wmma_bf16(a, t.bl[c], acc[c]);
  }
  SCHED_GROUP(0x0020, 20, 0);   // 20 VMEM reads first ...
  SCHED_GROUP(0x0008, 8, 0);    // ... then 8 WMMAs
}

__global__ __launch_bounds__(128)
void adj_attn(const float* __restrict__ adj,    // [B][N][N] fp32
              const __bf16* __restrict__ kThi,  // [B][64][N] bf16
              const __bf16* __restrict__ kTlo,  // [B][64][N] bf16
              const float* __restrict__ q,      // [B*N][64] fp32
              float* __restrict__ wpart) {      // [KSPLIT][B*N]
  const int lane = threadIdx.x & 31;
  const int wv   = threadIdx.x >> 5;
  const int csub = lane & 15;
  const int half = lane >> 4;
  const int b  = blockIdx.x >> 6;                  // 64 row-blocks per batch
  const int m0 = (blockIdx.x & 63) * 64 + wv * 16;
  const int ks = blockIdx.y;                       // K-split slice
  const int kt0   = ks * (NPTS / KSPLIT);
  const int ktEnd = kt0 + (NPTS / KSPLIT);
  const int mrow  = m0 + csub;
  const int koffA = half * 8;

  const float*  arow  = adj + (size_t)b * NPTS * NPTS + (size_t)mrow * NPTS + koffA;
  const __bf16* khrow = kThi + ((size_t)b * ADIM + csub) * NPTS + half * 8;
  const __bf16* klrow = kTlo + ((size_t)b * ADIM + csub) * NPTS + half * 8;

  v8f acc[4] = {};
  ATile tA, tB;
  a_load(tA, arow, khrow, klrow, kt0);
  // (NPTS/KSPLIT)/32 = 32 tiles -> 16 ping/pong pairs, no rotation copies
  for (int kt = kt0; kt < ktEnd; kt += 64) {
    a_load(tB, arow, khrow, klrow, kt + 32);
    a_compute(tA, acc);
    const int ktn = (kt + 64 < ktEnd) ? (kt + 64) : kt0;
    a_load(tA, arow, khrow, klrow, ktn);
    a_compute(tB, acc);
  }

  const float inv_scale = 0.125f;                // 1/sqrt(64)
#pragma unroll
  for (int r = 0; r < 8; ++r) {
    const int row = m0 + r + half * 8;
    float p = 0.f;
#pragma unroll
    for (int c = 0; c < 4; ++c)
      p += acc[c][r] * q[((size_t)b * NPTS + row) * ADIM + c * 16 + csub];
    p += __shfl_xor(p, 8, 16);                   // reduce within each half-wave
    p += __shfl_xor(p, 4, 16);
    p += __shfl_xor(p, 2, 16);
    p += __shfl_xor(p, 1, 16);
    if (csub == 0) wpart[(size_t)ks * BN + (size_t)b * NPTS + row] = p * inv_scale;
  }
}

// ------------------------------- sum K-split partials + softmax over bag dim
__global__ __launch_bounds__(256)
void softmax_bag(const float* __restrict__ wpart, float* __restrict__ wout) {
  __shared__ float red[256];
  const int t = threadIdx.x;
  const size_t base = (size_t)blockIdx.x * NPTS;
  float vals[16];
  float mx = -3.4e38f;
#pragma unroll
  for (int i = 0; i < 16; ++i) {
    const size_t idx = base + t + i * 256;
    float s = 0.f;
#pragma unroll
    for (int k = 0; k < KSPLIT; ++k) s += wpart[(size_t)k * BN + idx];
    vals[i] = s;
    mx = fmaxf(mx, s);
  }
  red[t] = mx; __syncthreads();
  for (int s = 128; s > 0; s >>= 1) { if (t < s) red[t] = fmaxf(red[t], red[t + s]); __syncthreads(); }
  mx = red[0]; __syncthreads();
  float sum = 0.f;
#pragma unroll
  for (int i = 0; i < 16; ++i) { vals[i] = __expf(vals[i] - mx); sum += vals[i]; }
  red[t] = sum; __syncthreads();
  for (int s = 128; s > 0; s >>= 1) { if (t < s) red[t] += red[t + s]; __syncthreads(); }
  const float inv = 1.f / red[0];
#pragma unroll
  for (int i = 0; i < 16; ++i) wout[base + t + i * 256] = vals[i] * inv;
}

// ---------------------------------------------------------- out = w * v
__global__ __launch_bounds__(256)
void scale_v(const float* __restrict__ v, const float* __restrict__ w,
             float* __restrict__ out) {
  const size_t i = (size_t)blockIdx.x * blockDim.x + threadIdx.x;  // BN*128
  const size_t row = i >> 7;                                       // 128 float4 per row
  v4f vv = *(const v4f*)(v + i * 4);
  const float s = w[row];
  v4f o = vv * s;
  *(v4f*)(out + i * 4) = o;
}

// ---------------------------------------------------------------------------
extern "C" void kernel_launch(void* const* d_in, const int* in_sizes, int n_in,
                              void* d_out, int out_size, void* d_ws, size_t ws_size,
                              hipStream_t stream) {
  (void)in_sizes; (void)n_in; (void)out_size; (void)ws_size;
  const float* x   = (const float*)d_in[0];   // (B,N,512)
  const float* adj = (const float*)d_in[1];   // (B,N,N)
  const float* Wqk = (const float*)d_in[2];   // (128,512)
  const float* Wv  = (const float*)d_in[3];   // (512,512)
  float* out = (float*)d_out;                 // (B,N,512)

  char* ws = (char*)d_ws;                     // ~57 MB total
  __bf16* xb    = (__bf16*)(ws);              // 16,777,216 B
  __bf16* wqkb  = (__bf16*)(ws + 16777216);   //    131,072 B
  __bf16* wvb   = (__bf16*)(ws + 16908288);   //    524,288 B
  float*  q     = (float*) (ws + 17432576);   //  4,194,304 B
  __bf16* kThi  = (__bf16*)(ws + 21626880);   //  2,097,152 B
  __bf16* kTlo  = (__bf16*)(ws + 23724032);   //  2,097,152 B
  float*  v     = (float*) (ws + 25821184);   // 33,554,432 B
  float*  wpart = (float*) (ws + 59375616);   //    262,144 B (KSPLIT slices)
  float*  wsm   = (float*) (ws + 59637760);   //     65,536 B

  cvt_f32_to_bf16_x4<<<(BN * DIN / 4 + 255) / 256, 256, 0, stream>>>(x, xb, BN * DIN / 4);
  cvt_f32_to_bf16_x4<<<(2 * ADIM * DIN / 4 + 255) / 256, 256, 0, stream>>>(Wqk, wqkb, 2 * ADIM * DIN / 4);
  cvt_f32_to_bf16_x4<<<(DIN * DIN / 4 + 255) / 256, 256, 0, stream>>>(Wv, wvb, DIN * DIN / 4);

  // q/k projection (E=128) and v projection (E=512)
  gemm_proj<<<dim3(BN / 64, 2), 128, 0, stream>>>(xb, wqkb, q, kThi, kTlo, nullptr, 2 * ADIM, 1);
  gemm_proj<<<dim3(BN / 64, DIN / 64), 128, 0, stream>>>(xb, wvb, nullptr, nullptr, nullptr, v, DIN, 0);

  // K-split partials of kagg = adj @ (k_hi + k_lo); wpart = q.kagg * inv_scale
  adj_attn<<<dim3(BATCH * (NPTS / 64), KSPLIT), 128, 0, stream>>>(adj, kThi, kTlo, q, wpart);

  softmax_bag<<<BATCH, 256, 0, stream>>>(wpart, wsm);
  scale_v<<<(BN * DIN / 4) / 256, 256, 0, stream>>>(v, wsm, out);
}
